// Attention_56736517980381
// MI455X (gfx1250) — matrix operand order repaired and checked by reading, not compiled, not run on hardware
//
#include <hip/hip_runtime.h>
#include <hip/hip_bf16.h>

// MI455X / gfx1250: wave32, WMMA 16x16x32 bf16->f32, double-buffered TDM
// tile staging (tensor_load_to_lds + s_wait_tensorcnt ping-pong).
// Pipeline:
//   convert x -> bf16; convert+transpose W_qkv, W_proj -> bf16 (so all GEMM /
//   attention tiles are row-major 2-D tiles -> Tensor Data Mover eligible);
//   WMMA GEMM qkv; extract V^T; WMMA flash attention; WMMA GEMM proj.

typedef __attribute__((ext_vector_type(16))) __bf16   v16bf;
typedef __attribute__((ext_vector_type(8)))  float    v8f;
typedef __attribute__((ext_vector_type(4)))  unsigned v4u;
typedef __attribute__((ext_vector_type(8)))  int      v8i;
typedef __attribute__((ext_vector_type(4)))  int      v4i;

union ABFrag { v4u u[2]; v16bf bf; };

#if __has_builtin(__builtin_amdgcn_tensor_load_to_lds)
#define HAVE_TDM 1
#else
#define HAVE_TDM 0
#endif

__device__ __forceinline__ __bf16 f32_to_bf16(float f) {
  unsigned u = __builtin_bit_cast(unsigned, f);
  u += 0x7FFFu + ((u >> 16) & 1u);  // round-to-nearest-even
  unsigned short hs = (unsigned short)(u >> 16);
  return __builtin_bit_cast(__bf16, hs);
}

// Wait until at most N tensor ops outstanding (immediate forms only).
__device__ __forceinline__ void wait_tensorcnt0() {
#if __has_builtin(__builtin_amdgcn_s_wait_tensorcnt)
  __builtin_amdgcn_s_wait_tensorcnt(0);
#else
  asm volatile("s_wait_tensorcnt 0x0" ::: "memory");
#endif
}
__device__ __forceinline__ void wait_tensorcnt2() {
#if __has_builtin(__builtin_amdgcn_s_wait_tensorcnt)
  __builtin_amdgcn_s_wait_tensorcnt(2);
#else
  asm volatile("s_wait_tensorcnt 0x2" ::: "memory");
#endif
}

// ---------------------------------------------------------------------------
// TDM: load a 2-D tile (tile_rows x 64 bf16) from global (row stride
// `stride0` elements) into LDS at byte offset lds_off, padding each 128B row
// with 16B so LDS row stride is 144B (=72 bf16, 16B-aligned fragment reads).
// D# layout per CDNA5 ISA ch.7/8 (group0 128b, group1 256b, groups2/3 zero).
// ---------------------------------------------------------------------------
__device__ __forceinline__ void tdm_load_tile64(unsigned lds_off,
                                                const __bf16* gptr,
                                                unsigned stride0_elems,
                                                unsigned tile_rows) {
#if HAVE_TDM
  unsigned long long ga = (unsigned long long)(uintptr_t)gptr;
  v4u g0;
  g0.x = 0x1u;                                   // count=1, user mode
  g0.y = lds_off;                                // D#.lds_addr
  g0.z = (unsigned)ga;                           // global_addr[31:0]
  g0.w = ((unsigned)(ga >> 32) & 0x01FFFFFFu)    // global_addr[56:32]
         | (2u << 30);                           // type = 2 ("image")
  v8i g1 = {};
  // data_size=1 (2B), pad_enable, pad_interval=4 (32 DW = one 64-elem row),
  // pad_amount=3 (4 DW = 16B)
  g1[0] = (int)((1u << 16) | (1u << 20) | (4u << 22) | (3u << 25));
  g1[1] = (int)((64u & 0xFFFFu) << 16);                       // tensor_dim0 lo16
  g1[2] = (int)((64u >> 16) | ((tile_rows & 0xFFFFu) << 16)); // d0 hi | dim1 lo
  g1[3] = (int)((tile_rows >> 16) | (64u << 16));             // dim1 hi | tile_dim0
  g1[4] = (int)(tile_rows & 0xFFFFu);                         // tile_dim1 (dim2=0)
  g1[5] = (int)stride0_elems;                                 // dim0_stride lo32
  g1[6] = 0;
  g1[7] = 0;
  v4i g2 = {};
  v4i g3 = {};
#if defined(__clang_major__) && __clang_major__ >= 23
  v8i g4 = {};
  __builtin_amdgcn_tensor_load_to_lds(g0, g1, g2, g3, g4, 0);
#else
  __builtin_amdgcn_tensor_load_to_lds(g0, g1, g2, g3, 0);
#endif
#else
  (void)lds_off; (void)gptr; (void)stride0_elems; (void)tile_rows;
#endif
}

// ---------------------------------------------------------------------------
// Elementwise fp32 -> bf16
// ---------------------------------------------------------------------------
__global__ void convert_f32_bf16(const float* __restrict__ src,
                                 __bf16* __restrict__ dst, int n) {
  int i = blockIdx.x * blockDim.x + threadIdx.x;
  int stride = gridDim.x * blockDim.x;
  for (; i < n; i += stride) dst[i] = f32_to_bf16(src[i]);
}

// fp32 src[R][C] -> bf16 dst[C][R]   (coalesced read along C)
__global__ void convert_transpose_f32_bf16(const float* __restrict__ src,
                                           __bf16* __restrict__ dst,
                                           int R, int C) {
  int c = blockIdx.x * blockDim.x + threadIdx.x;
  int r = blockIdx.y;
  if (c < C) dst[(size_t)c * R + r] = f32_to_bf16(src[(size_t)r * C + c]);
}

// vT[(b*16+h)*64+d][t] = qkv[(b*2048+t)*3072 + 2048 + h*64 + d]
__global__ void extract_vT(const __bf16* __restrict__ qkvb,
                           __bf16* __restrict__ vT) {
  int row = blockIdx.y;                         // 0..2047 = (b*16+h)*64 + d
  int t   = blockIdx.x * blockDim.x + threadIdx.x;
  int bh = row >> 6, d = row & 63;
  int b = bh >> 4, h = bh & 15;
  vT[(size_t)row * 2048 + t] =
      qkvb[(size_t)(b * 2048 + t) * 3072 + 2048 + h * 64 + d];
}

// ---------------------------------------------------------------------------
// WMMA GEMM: C[M,N] = A[M,K] @ BT[N,K]^T.  bf16 in, f32 accumulate.
// BM=64, BN=128, BK=64. 256 threads = 8 waves; wave w -> subtile row w>>1,
// 64-col half w&1 (4 subtiles, 8 WMMAs per K-stage). Double-buffered:
// wave 0 issues next stage's two TDM loads into the alternate buffer, then
// s_wait_tensorcnt<=2 guarantees the current buffer is complete -> the DMA
// for stage i+1 overlaps the WMMA work of stage i.
// ---------------------------------------------------------------------------
template <bool OUT_F32>
__global__ __launch_bounds__(256) void gemm_bf16_wmma(
    const __bf16* __restrict__ A, const __bf16* __restrict__ BT,
    void* __restrict__ Cout, int M, int N, int K) {
  constexpr int BM = 64, BN = 128, BK = 64;
  constexpr int LDT = BK + 8;  // 72 bf16 = 144B row stride
  __shared__ __bf16 As[2][BM * LDT];  // 2 x  9216 B
  __shared__ __bf16 Bs[2][BN * LDT];  // 2 x 18432 B

  const int tid  = threadIdx.x;
  const int lane = tid & 31;
  const int wave = tid >> 5;
  const int lmod = lane & 15;
  const int lhi  = lane >> 4;
  const int blockM = blockIdx.y * BM;
  const int blockN = blockIdx.x * BN;
  const int r = wave >> 1;   // subtile row (0..3)
  const int c = wave & 1;    // 64-col half
  const int kgA = lhi * 8;   // A-frag K group offset
  const int kbB = lhi * 8;   // B-frag K group offset

  v8f acc[4] = {};

  auto stage = [&](int p, int k0) {
#if HAVE_TDM
    if (wave == 0) {
      tdm_load_tile64((unsigned)(uintptr_t)(void*)&As[p][0],
                      A + (size_t)blockM * K + k0, (unsigned)K, BM);
      tdm_load_tile64((unsigned)(uintptr_t)(void*)&Bs[p][0],
                      BT + (size_t)blockN * K + k0, (unsigned)K, BN);
    }
#else
    for (int s = tid; s < (BM * BK) / 8; s += 256) {
      int row = s >> 3, col8 = s & 7;
      *(v4u*)(&As[p][0] + row * LDT + col8 * 8) =
          *(const v4u*)(A + (size_t)(blockM + row) * K + k0 + col8 * 8);
    }
    for (int s = tid; s < (BN * BK) / 8; s += 256) {
      int row = s >> 3, col8 = s & 7;
      *(v4u*)(&Bs[p][0] + row * LDT + col8 * 8) =
          *(const v4u*)(BT + (size_t)(blockN + row) * K + k0 + col8 * 8);
    }
#endif
  };

  const int nstages = K / BK;
  stage(0, 0);  // prologue fill of buffer 0

  for (int i = 0; i < nstages; ++i) {
    const int p = i & 1;
    if (i + 1 < nstages) {
      __syncthreads();               // readers of buf (1-p) from iter i-1 done
      stage(1 - p, (i + 1) * BK);    // DMA next stage (overlaps compute below)
#if HAVE_TDM
      if (wave == 0) wait_tensorcnt2();  // only the 2 new ops outstanding
#endif
    } else {
#if HAVE_TDM
      if (wave == 0) wait_tensorcnt0();
#endif
    }
    __syncthreads();                 // buffer p is ready for all waves

    ABFrag af[2];
    const __bf16* arow = &As[p][0] + (r * 16 + lmod) * LDT;
#pragma unroll
    for (int ks = 0; ks < 2; ++ks) {
      af[ks].u[0] = *(const v4u*)(arow + ks * 32 + kgA);
      af[ks].u[1] = *(const v4u*)(arow + ks * 32 + 16 + kgA);
    }
#pragma unroll
    for (int j = 0; j < 4; ++j) {
      const __bf16* brow = &Bs[p][0] + (c * 64 + j * 16 + lmod) * LDT;
#pragma unroll
      for (int ks = 0; ks < 2; ++ks) {
        ABFrag bfg;
        bfg.u[0] = *(const v4u*)(brow + ks * 32 + kbB);
        bfg.u[1] = *(const v4u*)(brow + ks * 32 + 16 + kbB);
        acc[j] = __builtin_amdgcn_wmma_f32_16x16x32_bf16(
            false, af[ks].bf, false, bfg.bf, (short)0, acc[j], false, false);
      }
    }
  }

#pragma unroll
  for (int j = 0; j < 4; ++j) {
    int coln = blockN + c * 64 + j * 16 + lmod;
#pragma unroll
    for (int v = 0; v < 8; ++v) {
      int rowm = blockM + r * 16 + v + 8 * lhi;
      if constexpr (OUT_F32)
        ((float*)Cout)[(size_t)rowm * N + coln] = acc[j][v];
      else
        ((__bf16*)Cout)[(size_t)rowm * N + coln] = f32_to_bf16(acc[j][v]);
    }
  }
}

// ---------------------------------------------------------------------------
// Flash attention. One (b,h) per blockIdx.y, 64 query rows per block,
// 4 waves x 16 rows. K tiles staged row-major, V tiles staged from the
// pre-transposed vT -- both via double-buffered TDM. Scale = 1/HEAD_DIM.
// ---------------------------------------------------------------------------
__global__ __launch_bounds__(128) void flash_attn_wmma(
    const __bf16* __restrict__ qkv,   // [B*T, 3072]
    const __bf16* __restrict__ vT,    // [B*H*64, T]
    __bf16* __restrict__ attn_out) {  // [B*T, 1024]
  constexpr int T = 2048, HD = 64, QLD = 3072, D = 1024;
  constexpr int LDT = HD + 8;  // 72
  __shared__ __bf16 Kb[2][64 * LDT];     // keys: row = token, col = dim
  __shared__ __bf16 Vb[2][64 * LDT];     // V^T : row = dim,   col = token
  __shared__ __bf16 Plds[4 * 16 * LDT];  // per-wave P strip

  const int tid  = threadIdx.x;
  const int lane = tid & 31;
  const int wave = tid >> 5;
  const int lmod = lane & 15;
  const int lhi  = lane >> 4;

  const int b = blockIdx.y >> 4, h = blockIdx.y & 15;
  const int q0 = blockIdx.x * 64 + wave * 16;

  const __bf16* base = qkv + (size_t)b * T * QLD + h * HD;
  const __bf16* Qp = base;
  const __bf16* Kp = base + D;
  const __bf16* vTbh = vT + (size_t)(b * 16 + h) * HD * T;

  ABFrag qf[2];
  {
    const __bf16* qrow = Qp + (size_t)(q0 + lmod) * QLD;
    const int kg = lhi * 8;
#pragma unroll
    for (int ks = 0; ks < 2; ++ks) {
      qf[ks].u[0] = *(const v4u*)(qrow + ks * 32 + kg);
      qf[ks].u[1] = *(const v4u*)(qrow + ks * 32 + 16 + kg);
    }
  }

  v8f o[4] = {};
  float mrow[8], lsum[8];
#pragma unroll
  for (int v = 0; v < 8; ++v) { mrow[v] = -1e30f; lsum[v] = 0.f; }
  const float pscale = 1.0f / 64.0f;

  auto stage = [&](int p, int kv0) {
#if HAVE_TDM
    if (wave == 0) {
      tdm_load_tile64((unsigned)(uintptr_t)(void*)&Kb[p][0],
                      Kp + (size_t)kv0 * QLD, QLD, 64);
      tdm_load_tile64((unsigned)(uintptr_t)(void*)&Vb[p][0],
                      vTbh + kv0, T, 64);
    }
#else
    for (int s = tid; s < 512; s += 128) {
      int row = s >> 3, col8 = s & 7;
      *(v4u*)(&Kb[p][0] + row * LDT + col8 * 8) =
          *(const v4u*)(Kp + (size_t)(kv0 + row) * QLD + col8 * 8);
      *(v4u*)(&Vb[p][0] + row * LDT + col8 * 8) =
          *(const v4u*)(vTbh + (size_t)row * T + kv0 + col8 * 8);
    }
#endif
  };

  constexpr int nstages = T / 64;  // 32
  stage(0, 0);

  for (int i = 0; i < nstages; ++i) {
    const int p = i & 1;
    if (i + 1 < nstages) {
      __syncthreads();
      stage(1 - p, (i + 1) * 64);   // DMA next KV block during this compute
#if HAVE_TDM
      if (wave == 0) wait_tensorcnt2();
#endif
    } else {
#if HAVE_TDM
      if (wave == 0) wait_tensorcnt0();
#endif
    }
    __syncthreads();

    // S = Q @ K^T
    v8f s4[4] = {};
#pragma unroll
    for (int j = 0; j < 4; ++j) {
      const __bf16* krow = &Kb[p][0] + (j * 16 + lmod) * LDT;
      const int kb = lhi * 8;
#pragma unroll
      for (int ks = 0; ks < 2; ++ks) {
        ABFrag kf;
        kf.u[0] = *(const v4u*)(krow + ks * 32 + kb);
        kf.u[1] = *(const v4u*)(krow + ks * 32 + 16 + kb);
        s4[j] = __builtin_amdgcn_wmma_f32_16x16x32_bf16(
            false, qf[ks].bf, false, kf.bf, (short)0, s4[j], false, false);
      }
    }

    // online softmax; row of element v is (v + 8*lhi)
#pragma unroll
    for (int v = 0; v < 8; ++v) {
      float mx = fmaxf(fmaxf(s4[0][v], s4[1][v]), fmaxf(s4[2][v], s4[3][v])) * pscale;
#pragma unroll
      for (int msk = 1; msk < 16; msk <<= 1)
        mx = fmaxf(mx, __shfl_xor(mx, msk, 32));
      float mnew = fmaxf(mrow[v], mx);
      float corr = __expf(mrow[v] - mnew);
      float rsum = 0.f;
#pragma unroll
      for (int j = 0; j < 4; ++j) {
        float pj = __expf(s4[j][v] * pscale - mnew);
        s4[j][v] = pj;
        rsum += pj;
      }
#pragma unroll
      for (int msk = 1; msk < 16; msk <<= 1) rsum += __shfl_xor(rsum, msk, 32);
      lsum[v] = lsum[v] * corr + rsum;
      mrow[v] = mnew;
#pragma unroll
      for (int j = 0; j < 4; ++j) o[j][v] *= corr;
    }

    // P: C-layout regs -> per-wave LDS strip -> A-layout fragments
    __bf16* P = Plds + wave * 16 * LDT;
#pragma unroll
    for (int j = 0; j < 4; ++j)
#pragma unroll
      for (int v = 0; v < 8; ++v)
        P[(v + 8 * lhi) * LDT + j * 16 + lmod] = f32_to_bf16(s4[j][v]);
    // LDS is in-order within a wave: no barrier needed for this strip.
    ABFrag pf[2];
    {
      const __bf16* prow = P + lmod * LDT;
      const int kg = lhi * 8;
#pragma unroll
      for (int ks = 0; ks < 2; ++ks) {
        pf[ks].u[0] = *(const v4u*)(prow + ks * 32 + kg);
        pf[ks].u[1] = *(const v4u*)(prow + ks * 32 + 16 + kg);
      }
    }

    // O += P @ V  (B fragments from V^T rows: contiguous tokens)
#pragma unroll
    for (int j = 0; j < 4; ++j) {
      const __bf16* vrow = &Vb[p][0] + (j * 16 + lmod) * LDT;
      const int kb = lhi * 8;
#pragma unroll
      for (int ks = 0; ks < 2; ++ks) {
        ABFrag vf;
        vf.u[0] = *(const v4u*)(vrow + ks * 32 + kb);
        vf.u[1] = *(const v4u*)(vrow + ks * 32 + 16 + kb);
        o[j] = __builtin_amdgcn_wmma_f32_16x16x32_bf16(
            false, pf[ks].bf, false, vf.bf, (short)0, o[j], false, false);
      }
    }
  }

  float inv[8];
#pragma unroll
  for (int v = 0; v < 8; ++v) inv[v] = 1.0f / lsum[v];
#pragma unroll
  for (int j = 0; j < 4; ++j)
#pragma unroll
    for (int v = 0; v < 8; ++v)
      attn_out[(size_t)(b * T + q0 + v + 8 * lhi) * D + h * HD + j * 16 + lmod] =
          f32_to_bf16(o[j][v] * inv[v]);
}

// ---------------------------------------------------------------------------
// Host side. Workspace layout (56 MiB total):
//   [ 0, 8M)  x_bf16      4096x1024
//   [ 8,14M)  WqkvT_bf16  3072x1024  (transposed)
//   [14,16M)  WprojT_bf16 1024x1024  (transposed)
//   [16,40M)  qkv_bf16    4096x3072
//   [40,48M)  attn_bf16   4096x1024
//   [48,56M)  vT_bf16     2048x2048  ([B*H*64][T])
// ---------------------------------------------------------------------------
extern "C" void kernel_launch(void* const* d_in, const int* in_sizes, int n_in,
                              void* d_out, int out_size, void* d_ws, size_t ws_size,
                              hipStream_t stream) {
  const float* x     = (const float*)d_in[0];
  const float* Wqkv  = (const float*)d_in[1];
  const float* Wproj = (const float*)d_in[2];
  float* out = (float*)d_out;

  char* ws = (char*)d_ws;
  __bf16* xb      = (__bf16*)(ws);
  __bf16* wqkvT   = (__bf16*)(ws + (8ull << 20));
  __bf16* wprojT  = (__bf16*)(ws + (14ull << 20));
  __bf16* qkvb    = (__bf16*)(ws + (16ull << 20));
  __bf16* attnb   = (__bf16*)(ws + (40ull << 20));
  __bf16* vT      = (__bf16*)(ws + (48ull << 20));

  const int M = 4096, D = 1024;

  convert_f32_bf16<<<512, 256, 0, stream>>>(x, xb, M * D);
  convert_transpose_f32_bf16<<<dim3(3072 / 256, D), 256, 0, stream>>>(
      Wqkv, wqkvT, D, 3 * D);
  convert_transpose_f32_bf16<<<dim3(1024 / 256, D), 256, 0, stream>>>(
      Wproj, wprojT, D, D);

  gemm_bf16_wmma<false><<<dim3(3072 / 128, M / 64), 256, 0, stream>>>(
      xb, wqkvT, (void*)qkvb, M, 3072, D);

  extract_vT<<<dim3(2048 / 256, 2048), 256, 0, stream>>>(qkvb, vT);

  flash_attn_wmma<<<dim3(2048 / 64, 32), 128, 0, stream>>>(qkvb, vT, attnb);

  gemm_bf16_wmma<true><<<dim3(1024 / 128, M / 64), 256, 0, stream>>>(
      attnb, wprojT, (void*)out, M, D, D);
}
